// RNN_65472481460516
// MI455X (gfx1250) — matrix operand order repaired and checked by reading, not compiled, not run on hardware
//
#include <hip/hip_runtime.h>
#include <math.h>

// ---------- WMMA fragment types (CDNA5 gfx1250, wave32) ----------
typedef __attribute__((ext_vector_type(16))) __bf16 v16bf;
typedef __attribute__((ext_vector_type(8)))  float  v8f;

union FragBF { v16bf v; uint4 q[2]; };

#if __has_builtin(__builtin_amdgcn_tanhf)
#define FAST_TANH(x) __builtin_amdgcn_tanhf(x)
#else
#define FAST_TANH(x) tanhf(x)
#endif

// packed f32 -> bf16x2 in one VALU op: D[15:0]=bf16(lo), D[31:16]=bf16(hi)
__device__ __forceinline__ unsigned pk_bf16(float lo, float hi) {
  unsigned r;
  asm("v_cvt_pk_bf16_f32 %0, %1, %2" : "=v"(r) : "v"(lo), "v"(hi));
  return r;
}

__device__ __forceinline__ uint4 pk8(float4 f0, float4 f1) {
  uint4 r;
  r.x = pk_bf16(f0.x, f0.y);
  r.y = pk_bf16(f0.z, f0.w);
  r.z = pk_bf16(f1.x, f1.y);
  r.w = pk_bf16(f1.z, f1.w);
  return r;
}

// CDNA5 async global->LDS copy (16B per lane), tracked by ASYNCcnt.
__device__ __forceinline__ void async_copy_b128(unsigned lds_byte_addr, const void* gsrc) {
  asm volatile("global_load_async_to_lds_b128 %0, %1, off"
               :: "v"(lds_byte_addr), "v"(gsrc)
               : "memory");
}
__device__ __forceinline__ void wait_async0() {
  asm volatile("s_wait_asynccnt 0x0" ::: "memory");
}

// =====================================================================
// Kernel 1: xp = x @ W1^T + b1   (M=32768, N=512, K=512)
// Block: 256 threads (8 waves). Block tile 64(M) x 128(N).
// Waves: 4 (M) x 2 (N); each wave computes 16x64 via 4 WMMA accumulators.
// xp is written straight into d_out's h_seq region (scan overwrites in place).
// =====================================================================
__global__ __launch_bounds__(256) void xp_gemm(const float* __restrict__ x,
                                               const float* __restrict__ W1,
                                               const float* __restrict__ bias1,
                                               float* __restrict__ out) {
  __shared__ __attribute__((aligned(16))) unsigned A_lds[64 * 16];   // 64 rows x 32 bf16
  __shared__ __attribute__((aligned(16))) unsigned B_lds[128 * 16];  // 128 rows x 32 bf16

  const int tid  = threadIdx.x;
  const int lane = tid & 31;
  const int wave = tid >> 5;
  const int wm   = wave >> 1;       // 0..3
  const int wn   = wave & 1;        // 0..1
  const int m0   = blockIdx.x * 64;
  const int n0   = blockIdx.y * 128;

  const int hi = lane >> 4;         // lane half
  const int lr = lane & 15;

  const int arow = tid >> 2, aks = (tid & 3) * 8;   // 64 rows, 8 floats each
  const int brow = tid >> 1, bks = (tid & 1) * 16;  // 128 rows, 16 floats each

  v8f acc[4] = {};

  for (int kk = 0; kk < 512; kk += 32) {
    const float4* ap = (const float4*)(x + (size_t)(m0 + arow) * 512 + kk + aks);
    uint4 apk = pk8(ap[0], ap[1]);
    const float4* bp = (const float4*)(W1 + (size_t)(n0 + brow) * 512 + kk + bks);
    uint4 bpk0 = pk8(bp[0], bp[1]);
    uint4 bpk1 = pk8(bp[2], bp[3]);

    *(uint4*)&A_lds[arow * 16 + aks / 2]     = apk;
    *(uint4*)&B_lds[brow * 16 + bks / 2]     = bpk0;
    *(uint4*)&B_lds[brow * 16 + bks / 2 + 4] = bpk1;
    __syncthreads();

    FragBF a;
    const unsigned* arowp = &A_lds[(wm * 16 + lr) * 16];
    a.q[0] = *(const uint4*)&arowp[hi * 4];
    a.q[1] = *(const uint4*)&arowp[hi * 4 + 8];

#pragma unroll
    for (int nt = 0; nt < 4; ++nt) {
      FragBF b;
      const unsigned* browp = &B_lds[(wn * 64 + nt * 16 + lr) * 16];
      b.q[0] = *(const uint4*)&browp[hi * 4];
      b.q[1] = *(const uint4*)&browp[hi * 4 + 8];
      acc[nt] = __builtin_amdgcn_wmma_f32_16x16x32_bf16(
          false, a.v, false, b.v, (short)0, acc[nt], false, false);
    }
    __syncthreads();
  }

#pragma unroll
  for (int nt = 0; nt < 4; ++nt) {
    const int n = n0 + wn * 64 + nt * 16 + lr;
    const float bb = bias1[n];
#pragma unroll
    for (int r = 0; r < 8; ++r) {
      const int m = m0 + wm * 16 + hi * 8 + r;
      out[(size_t)m * 512 + n] = acc[nt][r] + bb;
    }
  }
}

// =====================================================================
// Kernel 2: persistent recurrent scan.
// 16 workgroups x 256 threads. WG g owns columns [g*32, g*32+32).
// LDS: W2 slice 32x512 bf16 (32 KB, resident all 512 steps)
//      h_{t-1}  64x512 bf16 (64 KB, filled via global_load_async_to_lds_b128
//               from a double-buffered bf16 mirror of h kept in d_ws)
// 8 waves as 4(M=batch) x 2(N): each wave = one 16x16 tile, 16 WMMAs/step
// (K-loop fully unrolled so ds_load latency pipelines under the XDL pipe).
// Inter-WG handoff: device-scope release/acquire counter barrier in d_ws.
// =====================================================================
__global__ __launch_bounds__(256) void rnn_scan(const float* __restrict__ W2,
                                                const float* __restrict__ bias2,
                                                float* __restrict__ out,
                                                unsigned short* __restrict__ hbf,
                                                unsigned* __restrict__ cnt) {
  __shared__ __attribute__((aligned(16))) unsigned w2buf[32 * 256];  // 32 x 512 bf16
  __shared__ __attribute__((aligned(16))) unsigned hbuf[64 * 256];   // 64 x 512 bf16

  const int tid  = threadIdx.x;
  const int lane = tid & 31;
  const int wave = tid >> 5;
  const int tm   = wave & 3;    // batch tile 0..3
  const int tn   = wave >> 2;   // n tile 0..1
  const int n0   = blockIdx.x * 32;

  // ---- stage W2 slice once: rows n0..n0+31, all K (f32 -> bf16) ----
  {
    const int n  = tid >> 3;           // 0..31
    const int kb = (tid & 7) * 64;     // 0..448
    const float4* p = (const float4*)(W2 + (size_t)(n0 + n) * 512 + kb);
#pragma unroll
    for (int j = 0; j < 8; ++j) {
      *(uint4*)&w2buf[n * 256 + kb / 2 + j * 4] = pk8(p[2 * j], p[2 * j + 1]);
    }
  }
  __syncthreads();

  const int hi   = lane >> 4;
  const int lr   = lane & 15;
  const int arow = tm * 16 + lr;       // batch row for A frag
  const int ncol = tn * 16 + lr;       // local n for B frag / C col
  const int ng   = n0 + ncol;          // global n
  const float bb = bias2[ng];
  const unsigned hbuf_base = (unsigned)(uintptr_t)(&hbuf[0]);  // LDS byte address

  for (int t = 0; t < 512; ++t) {
    float* orow = out + (size_t)t * 32768;

    // ---- stage h_{t-1} into LDS ----
    if (t == 0) {
      const uint4 z = {0u, 0u, 0u, 0u};
#pragma unroll
      for (int i = 0; i < 8; ++i) {
        const int idx = i * 4096 + tid * 16;
        *(uint4*)&hbuf[idx >> 1]       = z;
        *(uint4*)&hbuf[(idx >> 1) + 4] = z;
      }
    } else {
      // async copy 64 KB bf16 mirror: 256 threads x 16 x 16B
      const char* hsrc = (const char*)(hbf + (size_t)((t - 1) & 1) * 32768);
#pragma unroll
      for (int i = 0; i < 16; ++i) {
        const int off = i * 4096 + tid * 16;   // bytes
        async_copy_b128(hbuf_base + (unsigned)off, hsrc + off);
      }
      wait_async0();
    }

    // prefetch this step's xp operands (consumed after the WMMA loop)
#pragma unroll
    for (int r = 0; r < 8; ++r) {
      __builtin_prefetch(&orow[(size_t)(tm * 16 + hi * 8 + r) * 512 + ng], 0, 3);
    }
    __syncthreads();

    // ---- 16x16 tile GEMM over K=512, fully unrolled (16 WMMAs) ----
    v8f acc = {};
    const unsigned* arowp = &hbuf[arow * 256];
    const unsigned* browp = &w2buf[ncol * 256];
#pragma unroll
    for (int ku = 0; ku < 256; ku += 16) {   // ku in dword units (= 2 bf16)
      FragBF a, b;
      a.q[0] = *(const uint4*)&arowp[ku + hi * 4];
      a.q[1] = *(const uint4*)&arowp[ku + hi * 4 + 8];
      b.q[0] = *(const uint4*)&browp[ku + hi * 4];
      b.q[1] = *(const uint4*)&browp[ku + hi * 4 + 8];
      acc = __builtin_amdgcn_wmma_f32_16x16x32_bf16(
          false, a.v, false, b.v, (short)0, acc, false, false);
    }

    // ---- fused epilogue: h_t = tanh(acc + xp_t + b2) ----
    // write f32 h_t in place over xp, plus bf16 mirror for next step's staging
    unsigned short* hrow = hbf + (size_t)(t & 1) * 32768;
#pragma unroll
    for (int r = 0; r < 8; ++r) {
      const int m = tm * 16 + hi * 8 + r;
      float v = FAST_TANH(acc[r] + orow[(size_t)m * 512 + ng] + bb);
      orow[(size_t)m * 512 + ng] = v;
      hrow[(size_t)m * 512 + ng] = (unsigned short)(pk_bf16(v, 0.0f) & 0xFFFFu);
      if (t == 511) out[(size_t)512 * 32768 + (size_t)m * 512 + ng] = v;  // h_last
    }

    // ---- inter-WG barrier: release stores, count arrivals, acquire ----
    __threadfence();
    __syncthreads();
    if (tid == 0) {
      __hip_atomic_fetch_add(cnt, 1u, __ATOMIC_RELEASE, __HIP_MEMORY_SCOPE_AGENT);
      const unsigned target = 16u * (unsigned)(t + 1);
      while (__hip_atomic_load(cnt, __ATOMIC_ACQUIRE, __HIP_MEMORY_SCOPE_AGENT) < target) {
        __builtin_amdgcn_s_sleep(16);
      }
    }
    __syncthreads();
    __threadfence();
  }
}

__global__ void init_cnt(unsigned* cnt) {
  if (threadIdx.x == 0) cnt[0] = 0u;
}

extern "C" void kernel_launch(void* const* d_in, const int* in_sizes, int n_in,
                              void* d_out, int out_size, void* d_ws, size_t ws_size,
                              hipStream_t stream) {
  const float* x   = (const float*)d_in[0];  // [512, 64, 512]
  const float* W1  = (const float*)d_in[1];  // [512, 512]
  const float* b1v = (const float*)d_in[2];  // [512]
  const float* W2  = (const float*)d_in[3];  // [512, 512]
  const float* b2v = (const float*)d_in[4];  // [512]
  float* out = (float*)d_out;                // h_seq [512,64,512] ++ h_last [64,512]

  // ws layout: [0,128K) two bf16 h mirrors (double buffer), then counter
  unsigned short* hbf = (unsigned short*)d_ws;
  unsigned* cnt = (unsigned*)((char*)d_ws + 2 * 64 * 512 * sizeof(unsigned short));

  init_cnt<<<1, 32, 0, stream>>>(cnt);

  dim3 g1(32768 / 64, 512 / 128);            // M tiles x N tiles
  xp_gemm<<<g1, 256, 0, stream>>>(x, W1, b1v, out);

  rnn_scan<<<16, 256, 0, stream>>>(W2, b2v, out, hbf, cnt);
}
